// ModulatedDCN2dv1_72524817760945
// MI455X (gfx1250) — matrix operand order repaired and checked by reading, not compiled, not run on hardware
//
#include <hip/hip_runtime.h>

#define Hd 128
#define Wd 128
#define HW 16384
#define Cc 128
#define Oc 128
#define Bc 4
#define KKc 9
#define NPIX 64
#define ROWP 136   // padded LDS row stride in bf16 elements (16B aligned, conflict-avoiding)

typedef __attribute__((ext_vector_type(16))) __bf16 v16bf;
typedef __attribute__((ext_vector_type(8)))  __bf16 v8bf;
typedef __attribute__((ext_vector_type(8)))  float  v8f;

static __device__ inline unsigned short f2bfu(float f) {
    unsigned u = __builtin_bit_cast(unsigned, f);
    u += 0x7FFFu + ((u >> 16) & 1u);          // round-to-nearest-even
    return (unsigned short)(u >> 16);
}
static __device__ inline __bf16 f2bf(float f) {
    return __builtin_bit_cast(__bf16, f2bfu(f));
}

// ---------------- Kernel 1: s = style @ (mod_weight/sqrt(512)).T + mod_bias ----------------
__global__ void style_mod_kernel(const float* __restrict__ style,
                                 const float* __restrict__ mod_weight,
                                 const float* __restrict__ mod_bias,
                                 float* __restrict__ s_out) {
    int b = blockIdx.x;
    int c = threadIdx.x;                      // 128 threads, one channel each
    const float* st = style + (size_t)b * 512;
    const float* mw = mod_weight + (size_t)c * 512;
    float acc = 0.f;
    for (int j = 0; j < 512; ++j) acc += st[j] * mw[j];
    s_out[b * Cc + c] = acc * 0.04419417382415922f + mod_bias[c];   // 1/sqrt(512)
}

// ---------------- Kernel 2: modulated + demodulated weight -> bf16 [b][k][o][c] ----------------
__global__ void modw_kernel(const float* __restrict__ weight,
                            const float* __restrict__ s_in,
                            __bf16* __restrict__ wbf) {
    int b = blockIdx.x >> 7;
    int o = blockIdx.x & 127;
    int c = threadIdx.x;                      // 128 threads
    float s = s_in[b * Cc + c];
    const float scale = 0.02946278254943948f; // 1/sqrt(C*K*K) = 1/sqrt(1152)
    const float* wp = weight + ((size_t)o * Cc + c) * KKc;
    float wv[9]; float sq = 0.f;
    #pragma unroll
    for (int k = 0; k < 9; ++k) { float v = scale * wp[k] * s; wv[k] = v; sq += v * v; }
    __shared__ float red[128];
    red[c] = sq;
    __syncthreads();
    for (int stp = 64; stp > 0; stp >>= 1) { if (c < stp) red[c] += red[c + stp]; __syncthreads(); }
    float demod = rsqrtf(red[0] + 1e-8f);
    #pragma unroll
    for (int k = 0; k < 9; ++k)
        wbf[(((size_t)(b * KKc + k) * Oc) + o) * Cc + c] = f2bf(wv[k] * demod);
}

// ---------------- Kernel 3: fused deform-sample + bf16 WMMA GEMM + activation ----------------
__global__ __launch_bounds__(256) void dcn_wmma_kernel(
    const float* __restrict__ inp,
    const float* __restrict__ offs,
    const float* __restrict__ mask,
    const __bf16* __restrict__ wbf,
    const float* __restrict__ act_bias,
    float* __restrict__ out) {

    __shared__ __align__(16) __bf16 wLds[Oc * ROWP];     // 128 x 136 bf16 (tap weight, row = o, cols = c)
    __shared__ __align__(16) __bf16 sLds[NPIX * ROWP];   // 64 x 136 bf16  (samples, row = pixel, cols = c)

    const int tid  = threadIdx.x;
    const int lane = tid & 31;
    const int wave = tid >> 5;                // 0..7 -> O-tile
    const int b     = blockIdx.x >> 8;        // 256 pixel-tiles per batch
    const int ptile = blockIdx.x & 255;
    const int p0    = ptile * NPIX;

    // sampling role: 4 channel-groups x 64 pixels
    const int pl = tid & 63;
    const int cg = tid >> 6;                  // 0..3, 32 channels each
    const int p  = p0 + pl;
    const int hh = p >> 7;
    const int ww = p & 127;

    // per-thread slice of the weight-tile async DMA: o = tid>>1, half-row = tid&1
    const int wo = tid >> 1, whf = tid & 1;
    const unsigned wLdsAddr = (unsigned)(uintptr_t)&wLds[wo * ROWP + whf * 64];
    const unsigned wGOffB   = (unsigned)((wo * Cc + whf * 64) * (int)sizeof(__bf16));

    v8f acc[4] = {};                          // 4 N-tiles of 16 pixels, f32 accum

    for (int k = 0; k < 9; ++k) {
        // ---- async DMA of tap weight 128x128 bf16 -> wLds (hidden behind sampling) ----
        // GVS mode: mem = SADDR(base) + VADDR(32b offset); LDS dest from per-lane VGPR.
        {
            const __bf16* wsrc = wbf + (((size_t)(b * KKc + k)) << 14);   // *128*128
            #pragma unroll
            for (int j = 0; j < 4; ++j) {
                unsigned laddr = wLdsAddr + j * 16;
                unsigned gaddr = wGOffB + j * 16;
                asm volatile("global_load_async_to_lds_b128 %0, %1, %2"
                             :
                             : "v"(laddr), "v"(gaddr), "s"(wsrc)
                             : "memory");
            }
        }
        // ---- deformable bilinear sampling for this tap -> sLds[pixel][channel] ----
        {
            const int ky = k / 3, kx = k % 3;
            float dy = offs[(((size_t)b * 18 + 2 * k    ) << 14) + p];
            float dx = offs[(((size_t)b * 18 + 2 * k + 1) << 14) + p];
            float m  = mask[(((size_t)b *  9 + k        ) << 14) + p];
            float py = (float)(hh + ky - 1) + dy;
            float px = (float)(ww + kx - 1) + dx;
            float y0f = floorf(py), x0f = floorf(px);
            int   y0 = (int)y0f,   x0 = (int)x0f;
            float wy = py - y0f,   wx = px - x0f;
            int   y1 = y0 + 1,     x1 = x0 + 1;
            bool vy0 = (y0 >= 0) && (y0 < Hd), vy1 = (y1 >= 0) && (y1 < Hd);
            bool vx0 = (x0 >= 0) && (x0 < Wd), vx1 = (x1 >= 0) && (x1 < Wd);
            float w00 = (vy0 && vx0) ? (1.f - wy) * (1.f - wx) * m : 0.f;
            float w01 = (vy0 && vx1) ? (1.f - wy) * wx        * m : 0.f;
            float w10 = (vy1 && vx0) ? wy        * (1.f - wx) * m : 0.f;
            float w11 = (vy1 && vx1) ? wy        * wx         * m : 0.f;
            int y0c = min(max(y0, 0), Hd - 1), y1c = min(max(y1, 0), Hd - 1);
            int x0c = min(max(x0, 0), Wd - 1), x1c = min(max(x1, 0), Wd - 1);
            int i00 = y0c * Wd + x0c, i01 = y0c * Wd + x1c;
            int i10 = y1c * Wd + x0c, i11 = y1c * Wd + x1c;
            const float* ipb = inp + (((size_t)b * Cc + cg * 32) << 14);
            __bf16* srow = &sLds[pl * ROWP + cg * 32];
            #pragma unroll 4
            for (int c2 = 0; c2 < 32; c2 += 2) {
                const float* ip0 = ipb + ((size_t)c2 << 14);
                const float* ip1 = ip0 + HW;
                float v0 = w00 * ip0[i00] + w01 * ip0[i01] + w10 * ip0[i10] + w11 * ip0[i11];
                float v1 = w00 * ip1[i00] + w01 * ip1[i01] + w10 * ip1[i10] + w11 * ip1[i11];
                unsigned pk = (unsigned)f2bfu(v0) | ((unsigned)f2bfu(v1) << 16);
                *(unsigned*)(srow + c2) = pk;       // 4B-aligned packed store
            }
        }
        // weight DMA must have landed before any wave reads wLds
        asm volatile("s_wait_asynccnt 0" ::: "memory");
        __syncthreads();
        // ---- WMMA: wave owns O rows [16*wave, 16*wave+16), all 4 pixel tiles ----
        {
            const int m  = 16 * wave + (lane & 15);
            const int kb = (lane < 16) ? 0 : 8;     // A layout: K {0..7,16..23} / {8..15,24..31}
            const int nb = (lane < 16) ? 0 : 8;     // B layout: K {0..7,16..23} / {8..15,24..31}
            #pragma unroll
            for (int kk = 0; kk < 4; ++kk) {        // K = 128 channels in chunks of 32
                const __bf16* ap = &wLds[m * ROWP + kk * 32 + kb];
                v8bf alo = *(const v8bf*)ap;
                v8bf ahi = *(const v8bf*)(ap + 16);
                v16bf a;
                #pragma unroll
                for (int i = 0; i < 8; ++i) { a[i] = alo[i]; a[i + 8] = ahi[i]; }
                #pragma unroll
                for (int t = 0; t < 4; ++t) {
                    const __bf16* bp = &sLds[(t * 16 + (lane & 15)) * ROWP + kk * 32 + nb];
                    v8bf blo = *(const v8bf*)bp;
                    v8bf bhi = *(const v8bf*)(bp + 16);
                    v16bf bv;
                    #pragma unroll
                    for (int i = 0; i < 8; ++i) { bv[i] = blo[i]; bv[i + 8] = bhi[i]; }
                    acc[t] = __builtin_amdgcn_wmma_f32_16x16x32_bf16(
                        false, a, false, bv, (short)0, acc[t], false, false);
                }
            }
        }
        __syncthreads();
    }

    // ---- epilogue: bias + leaky_relu(0.2) * sqrt(2), per C/D VGPR layout ----
    {
        const int oB = 16 * wave + ((lane < 16) ? 0 : 8);
        const int n  = lane & 15;
        #pragma unroll
        for (int t = 0; t < 4; ++t) {
            int pp = p0 + t * 16 + n;
            #pragma unroll
            for (int r = 0; r < 8; ++r) {
                int o = oB + r;
                float x = acc[t][r] + act_bias[o];
                float y = (x > 0.f ? x : 0.2f * x) * 1.4142135623730951f;
                out[(((size_t)b * Oc + o) << 14) + pp] = y;
            }
        }
    }
}

extern "C" void kernel_launch(void* const* d_in, const int* in_sizes, int n_in,
                              void* d_out, int out_size, void* d_ws, size_t ws_size,
                              hipStream_t stream) {
    const float* input      = (const float*)d_in[0];
    const float* style      = (const float*)d_in[1];
    const float* offset     = (const float*)d_in[2];
    const float* maskp      = (const float*)d_in[3];
    const float* weight     = (const float*)d_in[4];
    const float* mod_weight = (const float*)d_in[5];
    const float* mod_bias   = (const float*)d_in[6];
    const float* act_bias   = (const float*)d_in[7];
    float* out = (float*)d_out;

    float*  s_ws = (float*)d_ws;                          // B*C f32 = 2 KB
    __bf16* wbf  = (__bf16*)((char*)d_ws + 4096);         // B*9*O*C bf16 = 1.18 MB

    style_mod_kernel<<<dim3(Bc), dim3(128), 0, stream>>>(style, mod_weight, mod_bias, s_ws);
    modw_kernel<<<dim3(Bc * Oc), dim3(128), 0, stream>>>(weight, s_ws, wbf);
    dcn_wmma_kernel<<<dim3(Bc * 256), dim3(256), 0, stream>>>(input, offset, maskp, wbf,
                                                              act_bias, out);
}
